// HeteroGNNIDS_3745211483048
// MI455X (gfx1250) — matrix operand order repaired and checked by reading, not compiled, not run on hardware
//
#include <hip/hip_runtime.h>
#include <hip/hip_bf16.h>

typedef __attribute__((ext_vector_type(16))) __bf16 v16bf;
typedef __attribute__((ext_vector_type(8)))  float  v8f;

constexpr int NP = 5;       // edge types
constexpr int NE = 400000;  // edges per type (divisible by 64)
constexpr int NN = 50000;   // nodes
constexpr float SLOPE = 0.2f;

__device__ __forceinline__ __bf16 f2bf(float f) {
  union { float f; unsigned u; } cv; cv.f = f;
  unsigned u = cv.u;
  unsigned r = (u + 0x7FFFu + ((u >> 16) & 1u)) >> 16;  // round-to-nearest-even
  unsigned short h = (unsigned short)r;
  __bf16 o;
  __builtin_memcpy(&o, &h, sizeof(h));
  return o;
}

__device__ __forceinline__ void atomicMaxF(float* addr, float val) {
  int cur = __float_as_int(*addr);
  while (__int_as_float(cur) < val) {
    int prev = atomicCAS((int*)addr, cur, __float_as_int(val));
    if (prev == cur) break;
    cur = prev;
  }
}

// -------- ea.mean(0): once, layer-invariant. One block per (type, feature). --------
__global__ void k_ea_mean(const float* __restrict__ ea, float* __restrict__ mean) {
  int p = blockIdx.x / 35, f = blockIdx.x % 35;
  const float* base = ea + (size_t)p * NE * 35 + f;
  float s = 0.f;
  for (int e = threadIdx.x; e < NE; e += blockDim.x) s += base[(size_t)e * 35];
  __shared__ float red[256];
  red[threadIdx.x] = s; __syncthreads();
  for (int o = 128; o > 0; o >>= 1) {
    if (threadIdx.x < o) red[threadIdx.x] += red[threadIdx.x + o];
    __syncthreads();
  }
  if (threadIdx.x == 0) mean[p * 35 + f] = red[0] * (1.f / NE);
}

// -------- per-layer: self-loop edge projection (mean @ we) and summed bias --------
__global__ void k_prep(const float* __restrict__ mean, const float* __restrict__ we,
                       const float* __restrict__ bias, float* __restrict__ slproj,
                       float* __restrict__ bsum, int fo) {
  int t = threadIdx.x;
  if (t < NP * 16) {
    int p = t >> 4, f = t & 15;
    float s = 0.f;
    if (f < fo)
      for (int k = 0; k < 35; ++k) s += mean[p * 35 + k] * we[(p * 35 + k) * fo + f];
    slproj[t] = s;
  }
  if (t >= 128 && t < 144) {
    int f = t - 128; float s = 0.f;
    if (f < fo) for (int p = 0; p < NP; ++p) s += bias[p * fo + f];
    bsum[f] = s;
  }
}

// -------- per-layer init: emax=-inf, den=0, acc seeded with summed bias --------
__global__ void k_init(float* __restrict__ emax, float* __restrict__ den,
                       float* __restrict__ acc, const float* __restrict__ bsum, int fo) {
  int i = blockIdx.x * blockDim.x + threadIdx.x;
  if (i < NP * NN) { emax[i] = -3.4e38f; den[i] = 0.f; }
  else {
    int j = i - NP * NN;
    if (j < NN * 16) { int f = j & 15; acc[j] = (f < fo) ? bsum[f] : 0.f; }
  }
}

// -------- node projections xl = h@wl, xr = h@wr (padded to stride 16) + self-loop logit --------
__global__ void k_node(const float* __restrict__ in, int istride, int fi, int fo,
                       const float* __restrict__ wl, const float* __restrict__ wr,
                       const float* __restrict__ att, const float* __restrict__ slproj,
                       float* __restrict__ xl, float* __restrict__ xr,
                       float* __restrict__ sl, float* __restrict__ emax) {
  int n = blockIdx.x * blockDim.x + threadIdx.x;
  if (n >= NN) return;
  float h[16];
  for (int k = 0; k < 16; ++k) h[k] = (k < fi) ? in[(size_t)n * istride + k] : 0.f;
  for (int p = 0; p < NP; ++p) {
    const float* wlp = wl + p * fi * fo;
    const float* wrp = wr + p * fi * fo;
    float* xlp = xl + ((size_t)p * NN + n) * 16;
    float* xrp = xr + ((size_t)p * NN + n) * 16;
    float lg = 0.f;
    for (int f = 0; f < 16; ++f) {
      float a = 0.f, b = 0.f;
      if (f < fo) {
        for (int k = 0; k < fi; ++k) { a += h[k] * wlp[k * fo + f]; b += h[k] * wrp[k * fo + f]; }
      }
      xlp[f] = a; xrp[f] = b;
      if (f < fo) {
        float m = a + b + slproj[p * 16 + f];
        m = (m > 0.f) ? m : SLOPE * m;
        lg += m * att[p * fo + f];
      }
    }
    sl[(size_t)p * NN + n] = lg;
    atomicMaxF(&emax[(size_t)p * NN + n], lg);
  }
}

// -------- pass A (edges): WMMA edge-feature projection fused with attention logits --------
// Per wave: 16-edge tile. A = 16x64 bf16 (K=35 zero-padded), B = we zero-padded to 64x16.
// Two chained v_wmma_f32_16x16x32_bf16 accumulate the 16x16 f32 projection; C layout
// (VGPR r = edge row, lane = feature col) feeds directly into the per-edge logit reduce.
__global__ void __launch_bounds__(128) k_edgeA(
    const float* __restrict__ ea, const int* __restrict__ ei,
    const float* __restrict__ we, const float* __restrict__ att,
    const float* __restrict__ xl, const float* __restrict__ xr,
    float* __restrict__ elog, float* __restrict__ emax, int fo) {
  __shared__ float sA[64 * 36];  // 64 edges x 35 feats (+1 pad vs bank conflicts)
  const int p = blockIdx.y;
  const int tile0 = blockIdx.x * 64;
  const float* eap = ea + ((size_t)p * NE + tile0) * 35;
  for (int i = threadIdx.x; i < 64 * 35; i += 128) {
    int le = i / 35, k = i - le * 35;
    sA[le * 36 + k] = eap[i];
  }
  __syncthreads();

  const int lane = threadIdx.x & 31;
  const int wave = threadIdx.x >> 5;
  const int hi   = lane >> 4;    // half-wave selects K sub-block
  const int lrow = lane & 15;    // A row / B col / C feature

  const float* myrow = &sA[(wave * 16 + lrow) * 36];
  v16bf A0, A1;
#pragma unroll
  for (int j = 0; j < 16; ++j) {
    // 16-bit A 16x32 layout: lanes0-15 VGPR0-3 K=0..7, VGPR4-7 K=16..23; lanes16-31 +8
    int v = j >> 1;
    int k = (v < 4 ? 2 * v : 16 + 2 * (v - 4)) + hi * 8 + (j & 1);
    A0[j] = f2bf(myrow[k]);                       // k <= 31 < 35, always valid
    int k1 = k + 32;
    A1[j] = f2bf(k1 < 35 ? myrow[k1] : 0.f);      // only K=32..34 valid
  }

  const float* wep = we + (size_t)p * 35 * fo;
  v16bf B0, B1;
#pragma unroll
  for (int j = 0; j < 16; ++j) {
    // B 32x16 layout: lanes0-15 K=0..7,16..23; lanes16-31 +8, N = lane&15
    int k = (j < 8 ? j : 16 + (j - 8)) + hi * 8;
    B0[j] = f2bf(lrow < fo ? wep[k * fo + lrow] : 0.f);
    int k1 = k + 32;
    B1[j] = f2bf((k1 < 35 && lrow < fo) ? wep[k1 * fo + lrow] : 0.f);
  }

  v8f c = {};
  c = __builtin_amdgcn_wmma_f32_16x16x32_bf16(false, A0, false, B0, (short)0, c, false, false);
  c = __builtin_amdgcn_wmma_f32_16x16x32_bf16(false, A1, false, B1, (short)0, c, false, false);

  const int f = lrow;
  const float av = (f < fo) ? att[p * fo + f] : 0.f;
  const int* sp = ei + ((size_t)p * 2 + 0) * NE;
  const int* dp = ei + ((size_t)p * 2 + 1) * NE;
#pragma unroll
  for (int r = 0; r < 8; ++r) {
    int eg = tile0 + wave * 16 + hi * 8 + r;  // C layout: lanes0-15 M=r, lanes16-31 M=8+r
    int s = sp[eg], d = dp[eg];
    float mm = c[r] + xl[((size_t)p * NN + s) * 16 + f] + xr[((size_t)p * NN + d) * 16 + f];
    mm = (mm > 0.f) ? mm : SLOPE * mm;                 // leaky_relu before attention dot
    float contrib = mm * av;
#pragma unroll
    for (int o = 8; o >= 1; o >>= 1) contrib += __shfl_xor(contrib, o, 32);  // 16-lane sum
    if (f == 0) {
      elog[(size_t)p * NE + eg] = contrib;
      atomicMaxF(&emax[(size_t)p * NN + d], contrib);
    }
  }
}

// -------- pass B (edges): ex = exp(logit - max), accumulate denominators --------
__global__ void k_edgeB(const int* __restrict__ ei, const float* __restrict__ emax,
                        float* __restrict__ elog, float* __restrict__ den) {
  int e = blockIdx.x * blockDim.x + threadIdx.x;
  int p = blockIdx.y;
  if (e >= NE) return;
  int d = ei[((size_t)p * 2 + 1) * NE + e];
  float ex = __expf(elog[(size_t)p * NE + e] - emax[(size_t)p * NN + d]);
  elog[(size_t)p * NE + e] = ex;
  atomicAdd(&den[(size_t)p * NN + d], ex);
}

__global__ void k_selfB(const float* __restrict__ emax, float* __restrict__ sl,
                        float* __restrict__ den) {
  int i = blockIdx.x * blockDim.x + threadIdx.x;
  if (i >= NP * NN) return;
  float ex = __expf(sl[i] - emax[i]);
  sl[i] = ex;
  den[i] += ex;  // unique (p,n) per thread; kernel ordering serializes vs edge pass
}

__global__ void k_selfC(const float* __restrict__ sl, const float* __restrict__ den,
                        const float* __restrict__ xl, float* __restrict__ acc, int fo) {
  int n = blockIdx.x * blockDim.x + threadIdx.x;
  if (n >= NN) return;
  for (int p = 0; p < NP; ++p) {
    float alpha = sl[(size_t)p * NN + n] / den[(size_t)p * NN + n];
    const float* x = xl + ((size_t)p * NN + n) * 16;
    for (int f = 0; f < fo; ++f) acc[(size_t)n * 16 + f] += alpha * x[f];
  }
}

// -------- pass C (edges): out[d] += alpha * xl[s]; atomics stay L2-resident --------
__global__ void k_edgeC(const int* __restrict__ ei, const float* __restrict__ den,
                        const float* __restrict__ elog, const float* __restrict__ xl,
                        float* __restrict__ acc, int fo) {
  int e = blockIdx.x * blockDim.x + threadIdx.x;
  int p = blockIdx.y;
  if (e >= NE) return;
  int s = ei[((size_t)p * 2 + 0) * NE + e];
  int d = ei[((size_t)p * 2 + 1) * NE + e];
  float alpha = elog[(size_t)p * NE + e] / den[(size_t)p * NN + d];
  const float* x = xl + ((size_t)p * NN + s) * 16;
  float* a = acc + (size_t)d * 16;
  for (int f = 0; f < fo; ++f) atomicAdd(&a[f], alpha * x[f]);
}

__global__ void k_final(const float* __restrict__ acc, float* __restrict__ out,
                        int ostride, int fo, int relu) {
  int n = blockIdx.x * blockDim.x + threadIdx.x;
  if (n >= NN) return;
  for (int f = 0; f < ostride; ++f) {
    float v = (f < fo) ? acc[(size_t)n * 16 + f] : 0.f;
    if (relu) v = v > 0.f ? v : 0.f;
    out[(size_t)n * ostride + f] = v;
  }
}

extern "C" void kernel_launch(void* const* d_in, const int* in_sizes, int n_in,
                              void* d_out, int out_size, void* d_ws, size_t ws_size,
                              hipStream_t stream) {
  const float* x  = (const float*)d_in[0];
  const float* ea = (const float*)d_in[1];
  const int*   ei = (const int*)d_in[2];

  char* w = (char*)d_ws;
  size_t off = 0;
  auto alloc = [&](size_t bytes) -> float* {
    float* pt = (float*)(w + off);
    off = (off + bytes + 255) & ~(size_t)255;
    return pt;
  };
  float* ea_mean = alloc((size_t)NP * 35 * 4);
  float* slproj  = alloc((size_t)NP * 16 * 4);
  float* bsum    = alloc(16 * 4);
  float* hA      = alloc((size_t)NN * 16 * 4);
  float* hB      = alloc((size_t)NN * 16 * 4);
  float* xl      = alloc((size_t)NP * NN * 16 * 4);
  float* xr      = alloc((size_t)NP * NN * 16 * 4);
  float* elog    = alloc((size_t)NP * NE * 4);
  float* sl      = alloc((size_t)NP * NN * 4);
  float* emax    = alloc((size_t)NP * NN * 4);
  float* den     = alloc((size_t)NP * NN * 4);
  float* acc     = alloc((size_t)NN * 16 * 4);

  k_ea_mean<<<dim3(NP * 35), 256, 0, stream>>>(ea, ea_mean);

  struct Lyr { int fi, fo, relu; const float* in; int istride; float* out; int ostride; };
  Lyr layers[6] = {
    {15,  8, 1, x,  15, hA, 16},           // e1
    { 8,  8, 1, hA, 16, hB, 16},           // e2
    { 8,  2, 0, hB, 16, hA, 16},           // e3
    { 2,  8, 1, hA, 16, hB, 16},           // d1
    { 8,  8, 1, hB, 16, hA, 16},           // d2
    { 8, 15, 0, hA, 16, (float*)d_out, 15} // d3
  };

  for (int li = 0; li < 6; ++li) {
    const Lyr& L = layers[li];
    const float* wl = (const float*)d_in[3 + li * 5 + 0];
    const float* wr = (const float*)d_in[3 + li * 5 + 1];
    const float* we = (const float*)d_in[3 + li * 5 + 2];
    const float* at = (const float*)d_in[3 + li * 5 + 3];
    const float* bb = (const float*)d_in[3 + li * 5 + 4];

    k_prep<<<1, 256, 0, stream>>>(ea_mean, we, bb, slproj, bsum, L.fo);
    int initN = NP * NN + NN * 16;
    k_init<<<(initN + 255) / 256, 256, 0, stream>>>(emax, den, acc, bsum, L.fo);
    k_node<<<(NN + 127) / 128, 128, 0, stream>>>(L.in, L.istride, L.fi, L.fo,
                                                 wl, wr, at, slproj, xl, xr, sl, emax);
    k_edgeA<<<dim3(NE / 64, NP), 128, 0, stream>>>(ea, ei, we, at, xl, xr, elog, emax, L.fo);
    k_edgeB<<<dim3((NE + 255) / 256, NP), 256, 0, stream>>>(ei, emax, elog, den);
    k_selfB<<<(NP * NN + 255) / 256, 256, 0, stream>>>(emax, sl, den);
    k_selfC<<<(NN + 255) / 256, 256, 0, stream>>>(sl, den, xl, acc, L.fo);
    k_edgeC<<<dim3((NE + 255) / 256, NP), 256, 0, stream>>>(ei, den, elog, xl, acc, L.fo);
    k_final<<<(NN + 255) / 256, 256, 0, stream>>>(acc, L.out, L.ostride, L.fo, L.relu);
  }
}
